// DynamicGNNBlock_1365799600613
// MI455X (gfx1250) — matrix operand order repaired and checked by reading, not compiled, not run on hardware
//
#include <hip/hip_runtime.h>
#include <hip/hip_bf16.h>
#include <float.h>

typedef __attribute__((ext_vector_type(16))) _Float16 v16h;
typedef __attribute__((ext_vector_type(8)))  _Float16 v8h;
typedef __attribute__((ext_vector_type(8)))  float    v8f;

#define Bb 256
#define Cc 256
#define Nn 81
#define NP 96   // N padded to 6 tiles of 16
#define Kk 9

// ---------------- WMMA helpers (CDNA5 16x16x32 f16 -> f32) ----------------
__device__ __forceinline__ v8f wmma_f32_f16(v16h a, v16h b, v8f c) {
  // (neg_a, A, neg_b, B, c_mod, C, reuse_a, reuse_b)
  return __builtin_amdgcn_wmma_f32_16x16x32_f16(false, a, false, b, (short)0, c,
                                                false, false);
}

union V16U { v16h v; v8h h[2]; };

// A fragment: 16x32 f16 tile from row-major memory (stride in halves, 16B-aligned
// rows). Lane l: row = l&15, hi = l>>4; halves 0..7 -> K=hi*8+j ; 8..15 -> K=16+hi*8+j.
// Two contiguous 16-byte loads per lane.
__device__ __forceinline__ v16h load_a_frag(const _Float16* base, int stride) {
  const int lane = threadIdx.x & 31;
  const int r = lane & 15, hi = lane >> 4;
  const _Float16* p = base + (size_t)r * stride + hi * 8;
  V16U u;
  u.h[0] = *(const v8h*)(p);
  u.h[1] = *(const v8h*)(p + 16);
  return u.v;
}

// B fragment from pre-packed weights: wpk[kt][ct][lane][16 halves] — one
// contiguous 32-byte per-lane load, coalesced across the wave.
__device__ __forceinline__ v16h load_b_pk(const _Float16* __restrict__ wpk,
                                          int kt, int ct) {
  const int lane = threadIdx.x & 31;
  return *(const v16h*)(wpk + (((size_t)kt * 16 + ct) * 32 + lane) * 16);
}

// ---------------- small utility kernels ----------------
__global__ void k_zero(float* p, int n) {
  int i = blockIdx.x * 256 + threadIdx.x;
  if (i < n) p[i] = 0.f;
}

// Pack f32 weight into f16 WMMA-B-fragment order:
// dst[((kt*16+ct)*32+lane)*16+j] = src[(ct*16+(lane&15))*stride + col_off + kt*32+(lane>>4)*8+j+(j>=8?8:0)]
__global__ void k_pack_w(const float* __restrict__ src, _Float16* __restrict__ dst,
                         int stride, int col_off) {
  int i = blockIdx.x * 256 + threadIdx.x;   // over Cc*Cc = 65536
  int j = i & 15, lane = (i >> 4) & 31, ct = (i >> 9) & 15, kt = i >> 13;
  int col = lane & 15, hi = lane >> 4;
  int o = ct * 16 + col;
  int k = kt * 32 + hi * 8 + j + (j >= 8 ? 8 : 0);
  dst[i] = (_Float16)src[(size_t)o * stride + col_off + k];
}

// ---------------- coalesced transpose+convert: x[b][c][n] f32 -> xt[b][n][c] f16
__global__ void __launch_bounds__(256)
k_xpose(const float* __restrict__ x, _Float16* __restrict__ xt) {
  __shared__ float tile[16][17];
  const int b = blockIdx.y, n0 = blockIdx.x * 16;
  const int tid = threadIdx.x;
  const int tlo = tid & 15, thi = tid >> 4;
  for (int c0 = 0; c0 < Cc; c0 += 16) {
    int n = n0 + tlo;                       // coalesced read along n
    int c = c0 + thi;
    float v = (n < Nn) ? x[((size_t)b * Cc + c) * Nn + n] : 0.f;
    tile[thi][tlo] = v;
    __syncthreads();
    // coalesced write along c
    xt[((size_t)b * NP + n0 + thi) * Cc + c0 + tlo] = (_Float16)tile[tlo][thi];
    __syncthreads();
  }
}

// ---------------- shared GEMM: t[b][n][o] = A[b][n][:]·Wpk + bias[o]
// + masked BN stats (sum, sumsq per channel o). Pure vmem -> WMMA.
__global__ void __launch_bounds__(128)
k_gemm_bias_stats(const _Float16* __restrict__ A, const _Float16* __restrict__ wpk,
                  const float* __restrict__ bias, float* __restrict__ tbuf,
                  float* __restrict__ sum, float* __restrict__ sumsq) {
  const int b = blockIdx.y, n0 = blockIdx.x * 16;
  const int tid = threadIdx.x, wv = tid >> 5;
  v8f acc[4] = {};
  const _Float16* arow = A + ((size_t)b * NP + n0) * Cc;
  for (int kt = 0; kt < 8; ++kt) {
    v16h a = load_a_frag(arow + kt * 32, Cc);
#pragma unroll
    for (int t = 0; t < 4; ++t) {
      v16h bf = load_b_pk(wpk, kt, wv * 4 + t);
      acc[t] = wmma_f32_f16(a, bf, acc[t]);
    }
  }
  const int lane = tid & 31, colL = lane & 15, hi = lane >> 4;
#pragma unroll
  for (int t = 0; t < 4; ++t) {
    int col = (wv * 4 + t) * 16 + colL;
    float bs = bias[col], s = 0.f, s2 = 0.f;
#pragma unroll
    for (int v = 0; v < 8; ++v) {
      int n = n0 + v + hi * 8;
      float val = acc[t][v] + bs;
      tbuf[((size_t)b * NP + n) * Cc + col] = val;
      if (n < Nn) { s += val; s2 += val * val; }
    }
    atomicAdd(&sum[col], s);
    atomicAdd(&sumsq[col], s2);
  }
}

// ---------------- BN finalize: scale/shift per channel ----------------
__global__ void k_bn_fin(const float* __restrict__ sum, const float* __restrict__ sumsq,
                         const float* __restrict__ gamma, const float* __restrict__ beta,
                         float* __restrict__ scale, float* __restrict__ shift,
                         float inv_cnt) {
  int o = threadIdx.x;
  float m = sum[o] * inv_cnt;
  float v = sumsq[o] * inv_cnt - m * m;
  float sc = gamma[o] * rsqrtf(v + 1e-5f);
  scale[o] = sc;
  shift[o] = beta[o] - m * sc;
}

// ---------------- apply BN + exact GELU -> feats f16 [b][NP][C], sq[b][NP]
__global__ void __launch_bounds__(256)
k_bn_gelu(const float* __restrict__ tbuf, const float* __restrict__ scale,
          const float* __restrict__ shift, _Float16* __restrict__ feats,
          float* __restrict__ sq) {
  __shared__ float red[256];
  const int n = blockIdx.x, b = blockIdx.y, o = threadIdx.x;
  float gv = 0.f;
  if (n < Nn) {
    float v = tbuf[((size_t)b * NP + n) * Cc + o] * scale[o] + shift[o];
    gv = 0.5f * v * (1.f + erff(v * 0.70710678118f));
  }
  feats[((size_t)b * NP + n) * Cc + o] = (_Float16)gv;
  red[o] = gv * gv;
  __syncthreads();
  for (int s = 128; s > 0; s >>= 1) {
    if (o < s) red[o] += red[o + s];
    __syncthreads();
  }
  if (o == 0) sq[(size_t)b * NP + n] = red[0];
}

// ---------------- kNN: per point, top-K smallest d2 (stable ties) ----------------
__global__ void __launch_bounds__(128)
k_knn(const _Float16* __restrict__ feats, const float* __restrict__ sq,
      int* __restrict__ idx) {
  __shared__ alignas(32) _Float16 F[Nn * Cc];
  __shared__ float SQ[Nn];
  const int b = blockIdx.x, tid = threadIdx.x;
  for (int e = tid; e < Nn * Cc / 8; e += 128) {   // b128 copies
    int n = e >> 5, c8 = (e & 31) * 8;
    *(v8h*)&F[n * Cc + c8] = *(const v8h*)&feats[((size_t)b * NP + n) * Cc + c8];
  }
  if (tid < Nn) SQ[tid] = sq[(size_t)b * NP + tid];
  __syncthreads();
  if (tid >= Nn) return;
  const int n = tid;
  float bd[Kk];
  int bi[Kk];
#pragma unroll
  for (int j = 0; j < Kk; ++j) { bd[j] = FLT_MAX; bi[j] = 0; }
  for (int m = 0; m < Nn; ++m) {
    float dot = 0.f;
    for (int c = 0; c < Cc; c += 8) {            // vectorized LDS reads (b128)
      v8h fa = *(const v8h*)&F[n * Cc + c];
      v8h fb = *(const v8h*)&F[m * Cc + c];
#pragma unroll
      for (int j = 0; j < 8; ++j) dot += (float)fa[j] * (float)fb[j];
    }
    float d2 = fmaxf(SQ[n] + SQ[m] - 2.f * dot, 0.f);
    if (d2 < bd[Kk - 1]) {                       // strict < keeps earlier m on ties
      int j = Kk - 1;
      while (j > 0 && d2 < bd[j - 1]) { bd[j] = bd[j - 1]; bi[j] = bi[j - 1]; --j; }
      bd[j] = d2; bi[j] = m;
    }
  }
#pragma unroll
  for (int j = 0; j < Kk; ++j) idx[((size_t)b * Nn + n) * Kk + j] = bi[j];
}

// ---------------- u = feats*W1a^T - feats*W1b^T + b1 ; g = feats*W1b^T ----------
__global__ void __launch_bounds__(128)
k_gemm_ug(const _Float16* __restrict__ feats, const _Float16* __restrict__ w1apk,
          const _Float16* __restrict__ w1bpk, const float* __restrict__ b1,
          _Float16* __restrict__ ubuf, _Float16* __restrict__ gbuf) {
  const int b = blockIdx.y, n0 = blockIdx.x * 16;
  const int tid = threadIdx.x, wv = tid >> 5;
  v8f accU[4] = {}, accG[4] = {};
  const _Float16* arow = feats + ((size_t)b * NP + n0) * Cc;
  for (int kt = 0; kt < 8; ++kt) {
    v16h a = load_a_frag(arow + kt * 32, Cc);
#pragma unroll
    for (int t = 0; t < 4; ++t) {
      v16h bf1 = load_b_pk(w1apk, kt, wv * 4 + t);
      accU[t] = wmma_f32_f16(a, bf1, accU[t]);
      v16h bf2 = load_b_pk(w1bpk, kt, wv * 4 + t);
      accG[t] = wmma_f32_f16(a, bf2, accG[t]);
    }
  }
  const int lane = tid & 31, colL = lane & 15, hi = lane >> 4;
#pragma unroll
  for (int t = 0; t < 4; ++t) {
    int col = (wv * 4 + t) * 16 + colL;
    float bs = b1[col];
#pragma unroll
    for (int v = 0; v < 8; ++v) {
      int n = n0 + v + hi * 8;
      size_t off = ((size_t)b * NP + n) * Cc + col;
      float gval = accG[t][v];
      ubuf[off] = (_Float16)(accU[t][v] - gval + bs);
      gbuf[off] = (_Float16)gval;
    }
  }
}

// -------- per-point edge block: h[k] = lrelu(u[n] + g[idx[k]]); msg = h*w2^T + b2; max_k
__global__ void __launch_bounds__(128)
k_edge_msg(const _Float16* __restrict__ ubuf, const _Float16* __restrict__ gbuf,
           const int* __restrict__ idx, const _Float16* __restrict__ w2pk,
           const float* __restrict__ b2, _Float16* __restrict__ nf) {
  __shared__ alignas(32) _Float16 Ah[16 * Cc];
  __shared__ int nbr[Kk];
  const int n = blockIdx.x, b = blockIdx.y;
  const int tid = threadIdx.x, wv = tid >> 5;
  if (tid < Kk) nbr[tid] = idx[((size_t)b * Nn + n) * Kk + tid];
  __syncthreads();
  const _Float16* urow = ubuf + ((size_t)b * NP + n) * Cc;
  // build h rows in v8h chunks: 16 rows x 32 chunks of 8 halves
  for (int e = tid; e < 16 * 32; e += 128) {
    int k = e >> 5, c8 = (e & 31) * 8;
    v8h hv;
#pragma unroll
    for (int j = 0; j < 8; ++j) hv[j] = (_Float16)0.f;
    if (k < Kk) {
      const _Float16* grow = gbuf + ((size_t)b * NP + nbr[k]) * Cc;
      v8h uu = *(const v8h*)(urow + c8);
      v8h gg = *(const v8h*)(grow + c8);
#pragma unroll
      for (int j = 0; j < 8; ++j) {
        float h = (float)uu[j] + (float)gg[j];
        h = (h >= 0.f) ? h : 0.2f * h;           // LeakyReLU(0.2)
        hv[j] = (_Float16)h;
      }
    }
    *(v8h*)(Ah + k * Cc + c8) = hv;
  }
  __syncthreads();
  if ((tid & 31) == 0) __builtin_prefetch(w2pk, 0, 0);
  v8f acc[4] = {};
  for (int kt = 0; kt < 8; ++kt) {
    v16h a = load_a_frag(Ah + kt * 32, Cc);      // ds_load_b128 x2
#pragma unroll
    for (int t = 0; t < 4; ++t) {
      v16h bf = load_b_pk(w2pk, kt, wv * 4 + t);
      acc[t] = wmma_f32_f16(a, bf, acc[t]);
    }
  }
  const int lane = tid & 31, colL = lane & 15, hi = lane >> 4;
#pragma unroll
  for (int t = 0; t < 4; ++t) {
    float m;
    if (hi == 0) {                               // rows 0..7 all valid (k<9)
      m = acc[t][0];
#pragma unroll
      for (int v = 1; v < 8; ++v) m = fmaxf(m, acc[t][v]);
    } else {                                     // rows 8..15: only row 8 valid
      m = acc[t][0];
    }
    float other = __shfl_xor(m, 16);
    m = fmaxf(m, other);
    if (hi == 0) {
      int col = (wv * 4 + t) * 16 + colL;
      nf[((size_t)b * NP + n) * Cc + col] = (_Float16)(m + b2[col]);
    }
  }
}

// ---------------- finalize: relu(bn2(t) + residual) -> out [B][C][N] ----------------
__global__ void __launch_bounds__(256)
k_final(const float* __restrict__ tbuf, const float* __restrict__ scale,
        const float* __restrict__ shift, const float* __restrict__ x,
        float* __restrict__ out) {
  const int n = blockIdx.x, b = blockIdx.y, o = threadIdx.x;
  float v = tbuf[((size_t)b * NP + n) * Cc + o] * scale[o] + shift[o];
  v += x[((size_t)b * Cc + o) * Nn + n];
  out[((size_t)b * Cc + o) * Nn + n] = fmaxf(v, 0.f);
}

// =====================================================================
extern "C" void kernel_launch(void* const* d_in, const int* in_sizes, int n_in,
                              void* d_out, int out_size, void* d_ws, size_t ws_size,
                              hipStream_t stream) {
  const float* x      = (const float*)d_in[0];
  const float* w_in   = (const float*)d_in[1];
  const float* b_in   = (const float*)d_in[2];
  const float* g_in   = (const float*)d_in[3];
  const float* be_in  = (const float*)d_in[4];
  const float* w1     = (const float*)d_in[5];
  const float* b1     = (const float*)d_in[6];
  const float* w2     = (const float*)d_in[7];
  const float* b2     = (const float*)d_in[8];
  const float* w_out  = (const float*)d_in[9];
  const float* b_out  = (const float*)d_in[10];
  const float* g_out  = (const float*)d_in[11];
  const float* be_out = (const float*)d_in[12];
  float* out = (float*)d_out;

  // -------- workspace carve-up --------
  char* base = (char*)d_ws;
  size_t off = 0;
  auto carve = [&](size_t bytes) -> void* {
    void* p = base + off;
    off = (off + bytes + 255) & ~(size_t)255;
    return p;
  };
  float*     tbuf  = (float*)carve((size_t)Bb * NP * Cc * 4);      // reused by both convs
  _Float16*  xt    = (_Float16*)carve((size_t)Bb * NP * Cc * 2);   // transposed input
  _Float16*  feats = (_Float16*)carve((size_t)Bb * NP * Cc * 2);
  _Float16*  ubuf  = (_Float16*)carve((size_t)Bb * NP * Cc * 2);
  _Float16*  gbuf  = (_Float16*)carve((size_t)Bb * NP * Cc * 2);
  _Float16*  nf    = (_Float16*)carve((size_t)Bb * NP * Cc * 2);
  float*     sq    = (float*)carve((size_t)Bb * NP * 4);
  int*       idx   = (int*)carve((size_t)Bb * Nn * Kk * 4);
  _Float16*  w_inP = (_Float16*)carve((size_t)Cc * Cc * 2);   // packed fragment order
  _Float16*  w1aP  = (_Float16*)carve((size_t)Cc * Cc * 2);
  _Float16*  w1bP  = (_Float16*)carve((size_t)Cc * Cc * 2);
  _Float16*  w2P   = (_Float16*)carve((size_t)Cc * Cc * 2);
  _Float16*  w_oP  = (_Float16*)carve((size_t)Cc * Cc * 2);
  float*     stats = (float*)carve((size_t)8 * Cc * 4);
  // stats layout: [sum1, sumsq1, sum2, sumsq2, scale1, shift1, scale2, shift2]
  float* sum1 = stats, *sumsq1 = stats + Cc, *sum2 = stats + 2*Cc, *sumsq2 = stats + 3*Cc;
  float* scale1 = stats + 4*Cc, *shift1 = stats + 5*Cc;
  float* scale2 = stats + 6*Cc, *shift2 = stats + 7*Cc;

  const float inv_cnt = 1.f / (float)(Bb * Nn);

  // 0) zero BN accumulators (every call, for determinism)
  k_zero<<<4, 256, 0, stream>>>(stats, 4 * Cc);

  // 1) pack weights into f16 WMMA-B fragment order; transpose x to [b][n][c] f16
  k_pack_w<<<Cc, 256, 0, stream>>>(w_in,  w_inP, Cc,     0);
  k_pack_w<<<Cc, 256, 0, stream>>>(w1,    w1aP,  2 * Cc, 0);
  k_pack_w<<<Cc, 256, 0, stream>>>(w1,    w1bP,  2 * Cc, Cc);
  k_pack_w<<<Cc, 256, 0, stream>>>(w2,    w2P,   Cc,     0);
  k_pack_w<<<Cc, 256, 0, stream>>>(w_out, w_oP,  Cc,     0);
  k_xpose<<<dim3(NP / 16, Bb), 256, 0, stream>>>(x, xt);

  // 2) conv_in GEMM (WMMA) + BN1 stats
  k_gemm_bias_stats<<<dim3(NP / 16, Bb), 128, 0, stream>>>(xt, w_inP, b_in, tbuf,
                                                           sum1, sumsq1);
  // 3) BN1 finalize
  k_bn_fin<<<1, Cc, 0, stream>>>(sum1, sumsq1, g_in, be_in, scale1, shift1, inv_cnt);
  // 4) BN + GELU -> feats(f16), per-point squared norms
  k_bn_gelu<<<dim3(NP, Bb), 256, 0, stream>>>(tbuf, scale1, shift1, feats, sq);
  // 5) kNN (K=9 incl. self)
  k_knn<<<Bb, 128, 0, stream>>>(feats, sq, idx);
  // 6) per-point linear parts of the edge MLP (WMMA): u, g
  k_gemm_ug<<<dim3(NP / 16, Bb), 128, 0, stream>>>(feats, w1aP, w1bP, b1, ubuf, gbuf);
  // 7) per-point edge block: gather + LeakyReLU + msg GEMM (WMMA) + max_k
  k_edge_msg<<<dim3(Nn, Bb), 128, 0, stream>>>(ubuf, gbuf, idx, w2P, b2, nf);
  // 8) conv_out GEMM (WMMA) + BN2 stats
  k_gemm_bias_stats<<<dim3(NP / 16, Bb), 128, 0, stream>>>(nf, w_oP, b_out, tbuf,
                                                           sum2, sumsq2);
  // 9) BN2 finalize
  k_bn_fin<<<1, Cc, 0, stream>>>(sum2, sumsq2, g_out, be_out, scale2, shift2, inv_cnt);
  // 10) residual + ReLU -> out [B][C][H][W]
  k_final<<<dim3(Nn, Bb), 256, 0, stream>>>(tbuf, scale2, shift2, x, out);
}
